// HingeLoss_27650999452612
// MI455X (gfx1250) — matrix operand order repaired and checked by reading, not compiled, not run on hardware
//
#include <hip/hip_runtime.h>

// ---------------------------------------------------------------------------
// Problem constants (match reference)
// ---------------------------------------------------------------------------
#define N1 8192
#define N2 8192
#define CDIM 128
#define HC 60
#define WC 80
#define HIMG 480   // HC*8
#define WIMG 640   // WC*8
#define NEG_SAMPLES 256
#define POS_MARGIN 1.0f
#define NEG_MARGIN 0.2f

#define CHUNK_COLS 128                       // columns staged per LDS chunk
#define CHUNK_BYTES (CHUNK_COLS * CDIM * 2)  // 32768 bytes
#define NCHUNK (N2 / CHUNK_COLS)             // 64
#define ATILE_BYTES 4096                     // 16 rows x 128 f16

static constexpr float THR  = 11.413708498984762f;  // 2*sqrt(32)+0.1
static constexpr float THR2 = THR * THR;

typedef _Float16 v16h __attribute__((ext_vector_type(16)));
typedef _Float16 v8h  __attribute__((ext_vector_type(8)));
typedef float    v8f  __attribute__((ext_vector_type(8)));
typedef unsigned v4u  __attribute__((ext_vector_type(4)));
typedef int      v8i  __attribute__((ext_vector_type(8)));
typedef int      v4i  __attribute__((ext_vector_type(4)));

#if __has_builtin(__builtin_amdgcn_tensor_load_to_lds)
#define HAVE_TDM 1
#else
#define HAVE_TDM 0
#endif

// Monotone float -> uint key (order-preserving)
__device__ __forceinline__ unsigned fkey(float x) {
  unsigned u = __float_as_uint(x);
  return (u & 0x80000000u) ? ~u : (u | 0x80000000u);
}

// ---------------------------------------------------------------------------
// gfx1250 async global->LDS copy (16B granules) + asynccnt wait.
// VDST VGPR carries the LDS byte address: for a flat pointer into __shared__,
// the LDS offset is the low 32 bits (LDS aperture keeps offset in addr[31:0]).
// ---------------------------------------------------------------------------
__device__ __forceinline__ void async_copy16(const void* g, void* lds) {
  unsigned laddr = (unsigned)(size_t)lds;
  unsigned long long gaddr = (unsigned long long)(size_t)g;
  asm volatile("global_load_async_to_lds_b128 %0, %1, off"
               :: "v"(laddr), "v"(gaddr)
               : "memory");
}

__device__ __forceinline__ void wait_async0() {
#if __has_builtin(__builtin_amdgcn_s_wait_asynccnt)
  __builtin_amdgcn_s_wait_asynccnt(0);
#else
  asm volatile("s_wait_asynccnt 0x0" ::: "memory");
#endif
}

__device__ __forceinline__ void wait_tensor0() {
#if __has_builtin(__builtin_amdgcn_s_wait_tensorcnt)
  __builtin_amdgcn_s_wait_tensorcnt(0);
#else
  asm volatile("s_wait_tensorcnt 0x0" ::: "memory");
#endif
}

// Stage one 32KB column chunk of kp2f16 into LDS (all 256 threads, 8x16B each)
__device__ __forceinline__ void stage_chunk(const char* __restrict__ gbase,
                                            char* __restrict__ lbase, int tid) {
#pragma unroll
  for (int s = 0; s < 8; ++s) {
    const int off = (tid + 256 * s) * 16;
    async_copy16(gbase + off, lbase + off);
  }
}

#if HAVE_TDM
// ---------------------------------------------------------------------------
// Tensor Data Mover: 1-D tile load, nelem 2-byte elements, global -> LDS.
// D# group0: count=1 | lds_addr | global_addr[56:0] | type=2.
// D# group1: data_size=1(2B), tensor_dim0=tile_dim0=nelem, stride=nelem.
// Groups 2/3 zero (<=2D tensor). Tracked by TENSORcnt.
// This toolchain declares the 6-arg builtin (extra int32x8, zero-filled).
// ---------------------------------------------------------------------------
__device__ __forceinline__ void tdm_load_1d(const void* g, void* lds, unsigned nelem) {
  unsigned long long ga = (unsigned long long)(size_t)g;
  unsigned la = (unsigned)(size_t)lds;
  v4u g0;
  g0[0] = 1u;                                            // count=1 (valid, user)
  g0[1] = la;                                            // lds_addr
  g0[2] = (unsigned)ga;                                  // global_addr[31:0]
  g0[3] = (unsigned)((ga >> 32) & 0x01FFFFFFu) | (2u << 30);  // [56:32] | type=2
  v8i g1;
  g1[0] = (int)(1u << 16);                               // data_size = 1 -> 2 bytes
  g1[1] = (int)((nelem & 0xFFFFu) << 16);                // tensor_dim0[15:0] @bit48
  g1[2] = (int)((nelem >> 16) | ((1u & 0xFFFFu) << 16)); // tensor_dim0 hi, tensor_dim1=1
  g1[3] = (int)((nelem & 0xFFFFu) << 16);                // tile_dim0 @bit112
  g1[4] = 0;                                             // tile_dim1=0, tile_dim2=0
  g1[5] = (int)nelem;                                    // tensor_dim0_stride lo
  g1[6] = 0;
  g1[7] = 0;
  v4i gz4 = {0, 0, 0, 0};
  v8i gz8 = {0, 0, 0, 0, 0, 0, 0, 0};
  __builtin_amdgcn_tensor_load_to_lds(g0, g1, gz4, gz4, gz8, 0);
}
#endif

// ---------------------------------------------------------------------------
// WMMA fragment helpers.
// A (16x32 f16): lane L holds row M=L%16; lanes 0-15 -> K kt*32+0..15,
//                lanes 16-31 -> K kt*32+16..31.
// B (32x16 f16): lane L holds column N=L%16; lanes 0-15 K kt*32+0..15,
//                lanes 16-31 K kt*32+16..31 (contiguous 32B per fragment).
// ---------------------------------------------------------------------------
__device__ __forceinline__ void load_a_frags(const _Float16* __restrict__ base,
                                             int rowBase, int lane, v16h A[4]) {
  const int m  = lane & 15;
  const int hi = lane >> 4;
  const _Float16* rp = base + (size_t)(rowBase + m) * CDIM;
#pragma unroll
  for (int kt = 0; kt < 4; ++kt) {
    const int k0 = kt * 32 + (hi ? 16 : 0);
    const int k1 = k0 + 8;
    v8h lo = *(const v8h*)(rp + k0);
    v8h hh = *(const v8h*)(rp + k1);
    v16h a;
#pragma unroll
    for (int i = 0; i < 8; ++i) { a[i] = lo[i]; a[i + 8] = hh[i]; }
    A[kt] = a;
  }
}

// B fragments from the staged LDS chunk: issue all 8 ds_load_b128 first so a
// single dscnt wait covers the tile, then run the 4 WMMAs back-to-back.
__device__ __forceinline__ v8f tile_acc_lds(const v16h A[4],
                                            const char* __restrict__ ldsB,
                                            int col0, int lane) {
  const int n  = lane & 15;
  const int hi = lane >> 4;
  const _Float16* cp = (const _Float16*)ldsB + (size_t)(col0 + n) * CDIM + (hi ? 16 : 0);
  v16h b[4];
#pragma unroll
  for (int kt = 0; kt < 4; ++kt) b[kt] = *(const v16h*)(cp + kt * 32);
  v8f acc = {};
#pragma unroll
  for (int kt = 0; kt < 4; ++kt) {
    acc = __builtin_amdgcn_wmma_f32_16x16x32_f16(
        /*neg_a=*/false, A[kt], /*neg_b=*/false, b[kt],
        /*c_mod=*/(short)0, acc, /*reuse_a=*/false, /*reuse_b=*/false);
  }
  return acc;
}

// ---------------------------------------------------------------------------
// Kernel 0: f32 -> f16 conversion of both descriptor matrices
// ---------------------------------------------------------------------------
__global__ void cvt_kernel(const float* __restrict__ a, const float* __restrict__ b,
                           _Float16* __restrict__ fa, _Float16* __restrict__ fb, int n) {
  int i = blockIdx.x * blockDim.x + threadIdx.x;
  if (i < n) {
    fa[i] = (_Float16)a[i];
    fb[i] = (_Float16)b[i];
  }
}

// ---------------------------------------------------------------------------
// Kernel 1: positive term. One wave (32 lanes) per row.
// ---------------------------------------------------------------------------
__global__ void pos_kernel(const float* __restrict__ w_kp1,
                           const float* __restrict__ kp1_desc,
                           const float* __restrict__ desc2,
                           float* __restrict__ posPartial) {
  __shared__ float wsum[8];
  const int lane = threadIdx.x & 31;
  const int wave = threadIdx.x >> 5;
  const int row  = blockIdx.x * 8 + wave;

  float y = w_kp1[2 * row + 0];
  float x = w_kp1[2 * row + 1];
  float py = y * ((float)(HC - 1) / (float)(HIMG - 1));
  float px = x * ((float)(WC - 1) / (float)(WIMG - 1));
  py = fminf(fmaxf(py, 0.0f), (float)(HC - 1));
  px = fminf(fmaxf(px, 0.0f), (float)(WC - 1));
  int y0 = min(max((int)floorf(py), 0), HC - 2);
  int x0 = min(max((int)floorf(px), 0), WC - 2);
  float wy = py - (float)y0;
  float wx = px - (float)x0;
  float w00 = (1.0f - wy) * (1.0f - wx);
  float w01 = (1.0f - wy) * wx;
  float w10 = wy * (1.0f - wx);
  float w11 = wy * wx;

  float n2 = 0.0f, dt = 0.0f;
  const int c0 = lane * 4;
#pragma unroll
  for (int cc = 0; cc < 4; ++cc) {
    int c = c0 + cc;
    size_t base = (size_t)c * (HC * WC) + (size_t)y0 * WC + x0;
    float v00 = desc2[base];
    float v01 = desc2[base + 1];
    float v10 = desc2[base + WC];
    float v11 = desc2[base + WC + 1];
    float o = v00 * w00 + v01 * w01 + v10 * w10 + v11 * w11;
    n2 += o * o;
    dt += o * kp1_desc[(size_t)row * CDIM + c];
  }
#pragma unroll
  for (int m = 1; m < 32; m <<= 1) {
    n2 += __shfl_xor(n2, m);
    dt += __shfl_xor(dt, m);
  }
  if (lane == 0) {
    float pd = dt / fmaxf(sqrtf(n2), 1e-12f);
    wsum[wave] = fmaxf(POS_MARGIN - pd, 0.0f);
  }
  __syncthreads();
  if (threadIdx.x == 0) {
    float s = 0.0f;
    for (int w = 0; w < 8; ++w) s += wsum[w];
    posPartial[blockIdx.x] = s;
  }
}

// ---------------------------------------------------------------------------
// Fused WMMA GEMM pass with async double-buffered LDS B-feed and a TDM
// (tensor_load_to_lds) A-tile load.
// MODE 0: per-row 4096-bin histogram of score keys -> top-256 boundary bin.
// MODE 1: sum relu(x-0.2) above boundary (averaged within the boundary bin).
// LDS: [2 x 32KB B ring][4KB A tile][MODE0: 128KB histogram]
// ---------------------------------------------------------------------------
template <int MODE>
__global__ void gemm_pass(const _Float16* __restrict__ kp1f,
                          const _Float16* __restrict__ kp2f,
                          const float* __restrict__ w_kp1,
                          const float* __restrict__ kp2,
                          unsigned* __restrict__ rowBin,
                          unsigned* __restrict__ rowNeed,
                          float* __restrict__ negPartial) {
  extern __shared__ char smem[];
  char* Bring = smem;                                    // 2 * CHUNK_BYTES
  char* Atile = smem + 2 * CHUNK_BYTES;                  // ATILE_BYTES
  unsigned* hist = (unsigned*)(smem + 2 * CHUNK_BYTES + ATILE_BYTES); // MODE 0

  __shared__ float    redF[8][16];
  __shared__ float    redB[8][16];
  __shared__ unsigned redC[8][16];
  __shared__ float    rloss[16];

  const int tid   = threadIdx.x;
  const int lane  = tid & 31;
  const int wave  = tid >> 5;
  const int rowBase = blockIdx.x * 16;
  const int dHalf = lane >> 4;  // C/D mapping: acc[r] is row r + 8*dHalf

  // Kick off the first async B chunk and the TDM A-tile load immediately.
  stage_chunk((const char*)kp2f, Bring, tid);
#if HAVE_TDM
  if (wave == 0)
    tdm_load_1d(kp1f + (size_t)rowBase * CDIM, Atile, 16 * CDIM);
#endif

  if (MODE == 0) {
    for (int i = tid; i < 8 * 4096; i += 256) hist[i] = 0;
  }

  float wy[8], wx[8];
  unsigned rb[8];
#pragma unroll
  for (int r = 0; r < 8; ++r) {
    int m = rowBase + r + 8 * dHalf;
    wy[r] = w_kp1[2 * m + 0];
    wx[r] = w_kp1[2 * m + 1];
    if (MODE == 1) rb[r] = rowBin[m];
  }

  v16h A[4];
#if HAVE_TDM
  if (wave == 0) wait_tensor0();
  __syncthreads();  // publish A tile to all waves
  load_a_frags((const _Float16*)Atile, 0, lane, A);
#else
  load_a_frags(kp1f, rowBase, lane, A);
#endif

  float sF[8], sB[8];
  unsigned cB[8];
#pragma unroll
  for (int r = 0; r < 8; ++r) { sF[r] = 0.0f; sB[r] = 0.0f; cB[r] = 0u; }
  const unsigned addv = dHalf ? (1u << 16) : 1u;

  for (int c = 0; c < NCHUNK; ++c) {
    wait_async0();        // my share of chunk c is in LDS
    __syncthreads();      // all waves staged chunk c AND finished chunk c-1
    if (c + 1 < NCHUNK)
      stage_chunk((const char*)kp2f + (size_t)(c + 1) * CHUNK_BYTES,
                  Bring + ((c + 1) & 1) * CHUNK_BYTES, tid);

    // Each wave computes one 16-column tile of this chunk.
    const int j0 = (c * 8 + wave) * 16;
    v8f acc = tile_acc_lds(A, Bring + (c & 1) * CHUNK_BYTES, wave * 16, lane);

    const int j = j0 + (lane & 15);
    const float ky = kp2[2 * j + 0];
    const float kx = kp2[2 * j + 1];
#pragma unroll
    for (int r = 0; r < 8; ++r) {
      float dy = wy[r] - ky, dx = wx[r] - kx;
      float sc = acc[r] - ((dy * dy + dx * dx) <= THR2 ? 5.0f : 0.0f);
      unsigned bin = fkey(sc) >> 20;
      if (MODE == 0) {
        atomicAdd(&hist[bin * 8 + r], addv);
      } else {
        float h = fmaxf(sc - NEG_MARGIN, 0.0f);
        if (bin > rb[r]) {
          sF[r] += h;
        } else if (bin == rb[r]) {
          sB[r] += h;
          cB[r] += 1u;
        }
      }
    }
  }
  __syncthreads();

  if (MODE == 0) {
    // Descending scan: find the top-256 boundary bin per row.
    if (tid < 16) {
      const int r = tid & 7, half = tid >> 3;
      unsigned cum = 0, binSel = 0, need = NEG_SAMPLES;
      for (int b = 4095; b >= 0; --b) {
        unsigned cnt = (hist[b * 8 + r] >> (16 * half)) & 0xFFFFu;
        if (cum + cnt >= (unsigned)NEG_SAMPLES) {
          binSel = (unsigned)b;
          need = NEG_SAMPLES - cum;
          break;
        }
        cum += cnt;
      }
      int m = rowBase + r + 8 * half;
      rowBin[m]  = binSel;
      rowNeed[m] = need;
    }
  } else {
    // Deterministic reduction: shuffle within 16-lane halves, then fixed order.
#pragma unroll
    for (int m = 1; m < 16; m <<= 1) {
#pragma unroll
      for (int r = 0; r < 8; ++r) {
        sF[r] += __shfl_xor(sF[r], m);
        sB[r] += __shfl_xor(sB[r], m);
        cB[r] += (unsigned)__shfl_xor((int)cB[r], m);
      }
    }
    if ((lane & 15) == 0) {
#pragma unroll
      for (int r = 0; r < 8; ++r) {
        redF[wave][r + 8 * dHalf] = sF[r];
        redB[wave][r + 8 * dHalf] = sB[r];
        redC[wave][r + 8 * dHalf] = cB[r];
      }
    }
    __syncthreads();
    if (tid < 16) {
      float F = 0.0f, B = 0.0f;
      unsigned C = 0u;
      for (int w = 0; w < 8; ++w) {
        F += redF[w][tid];
        B += redB[w][tid];
        C += redC[w][tid];
      }
      const unsigned need = rowNeed[rowBase + tid];
      float scale = (C > 0u) ? fminf((float)need, (float)C) / (float)C : 0.0f;
      rloss[tid] = F + B * scale;
    }
    __syncthreads();
    if (tid == 0) {
      float s = 0.0f;
      for (int i = 0; i < 16; ++i) s += rloss[i];
      negPartial[blockIdx.x] = s;
    }
  }
}

// ---------------------------------------------------------------------------
// Kernel 4: deterministic fixed-order scalar combine.
// ---------------------------------------------------------------------------
__global__ void final_reduce(const float* __restrict__ posPartial,
                             const float* __restrict__ negPartial,
                             float* __restrict__ out) {
  if (threadIdx.x == 0 && blockIdx.x == 0) {
    float p = 0.0f;
    for (int i = 0; i < N1 / 8; ++i) p += posPartial[i];
    float ng = 0.0f;
    for (int i = 0; i < N1 / 16; ++i) ng += negPartial[i];
    float loss = (p * ((float)NEG_SAMPLES / 3.0f) + ng) *
                 (1.0f / ((float)N1 * (float)NEG_SAMPLES));
    out[0] = loss;
  }
}

// ---------------------------------------------------------------------------
// Host-side launcher
// ---------------------------------------------------------------------------
extern "C" void kernel_launch(void* const* d_in, const int* in_sizes, int n_in,
                              void* d_out, int out_size, void* d_ws, size_t ws_size,
                              hipStream_t stream) {
  // Inputs in setup_inputs() order:
  // 0: kp1 (unused)  1: w_kp1  2: kp2  3: kp1_desc  4: kp2_desc  5: desc2
  const float* w_kp1    = (const float*)d_in[1];
  const float* kp2      = (const float*)d_in[2];
  const float* kp1_desc = (const float*)d_in[3];
  const float* kp2_desc = (const float*)d_in[4];
  const float* desc2    = (const float*)d_in[5];
  float* out = (float*)d_out;

  // Workspace layout
  char* ws = (char*)d_ws;
  size_t off = 0;
  _Float16* kp1f = (_Float16*)(ws + off); off += (size_t)N1 * CDIM * sizeof(_Float16);
  _Float16* kp2f = (_Float16*)(ws + off); off += (size_t)N2 * CDIM * sizeof(_Float16);
  unsigned* rowBin  = (unsigned*)(ws + off); off += (size_t)N1 * sizeof(unsigned);
  unsigned* rowNeed = (unsigned*)(ws + off); off += (size_t)N1 * sizeof(unsigned);
  float* posPartial = (float*)(ws + off); off += (size_t)(N1 / 8) * sizeof(float);
  float* negPartial = (float*)(ws + off); off += (size_t)(N1 / 16) * sizeof(float);
  (void)ws_size; (void)in_sizes; (void)n_in; (void)out_size;

  const int nElem = N1 * CDIM;  // == N2 * CDIM
  cvt_kernel<<<(nElem + 255) / 256, 256, 0, stream>>>(kp1_desc, kp2_desc, kp1f, kp2f, nElem);
  pos_kernel<<<N1 / 8, 256, 0, stream>>>(w_kp1, kp1_desc, desc2, posPartial);

  const size_t lds_hist = 2 * CHUNK_BYTES + ATILE_BYTES + 8 * 4096 * sizeof(unsigned);
  const size_t lds_acc  = 2 * CHUNK_BYTES + ATILE_BYTES;
  gemm_pass<0><<<N1 / 16, 256, lds_hist, stream>>>(
      kp1f, kp2f, w_kp1, kp2, rowBin, rowNeed, nullptr);
  gemm_pass<1><<<N1 / 16, 256, lds_acc, stream>>>(
      kp1f, kp2f, w_kp1, kp2, rowBin, rowNeed, negPartial);
  final_reduce<<<1, 64, 0, stream>>>(posPartial, negPartial, out);
}
